// ExpertParallelMoE_2765958938932
// MI455X (gfx1250) — matrix operand order repaired and checked by reading, not compiled, not run on hardware
//
#include <hip/hip_runtime.h>
#include <math.h>

// ---------------------------------------------------------------------------
// MoE (top-2 of 8 experts, GELU MLP 768->3072->768) for gfx1250 (MI455X)
// bf16 WMMA path: v_wmma_f32_16x16x32_bf16, fp32 accumulate.
// Double-buffered LDS pipeline: one barrier per K-chunk.
// ---------------------------------------------------------------------------

typedef __attribute__((ext_vector_type(16))) __bf16 v16bf;
typedef __attribute__((ext_vector_type(8)))  float  v8f;

#define NTOK     2048
#define DMODEL   768
#define DFF      3072
#define NEXP     8
#define EXPCAP   2048          // max slots per expert (each token adds <=1)
#define MTILE    128           // block M tile (row-padding granularity)
#define NTILE    64            // block N tile
#define KSTEP    32
#define ASTRIDE  40            // As row stride (bf16 elems), keeps 16B alignment
#define BSTRIDE  40            // Bs col stride (bf16 elems)
#define ROWCAP   (NTOK*2 + NEXP*MTILE)   // 4096 + 1024 = 5120 compact rows max

struct bf16x16_bits { uint4 lo, hi; };

__device__ inline float gelu_tanh_fast(float v) {
    // tanh approx GELU, branchless: tanh(c) = 1 - 2/(exp(2c)+1)
    float c = 0.7978845608028654f * (v + 0.044715f * v * v * v);
    float u = __builtin_amdgcn_exp2f(2.885390081777927f * c); // exp(2c)
    float th = 1.0f - 2.0f * __builtin_amdgcn_rcpf(u + 1.0f);
    return 0.5f * v * (1.0f + th);
}

// ---------------------------------------------------------------------------
// K0: zero output + expert counters
// ---------------------------------------------------------------------------
__global__ void moe_zero_kernel(float* __restrict__ out, int n, int* __restrict__ cnt) {
    int i = blockIdx.x * blockDim.x + threadIdx.x;
    if (i < n) out[i] = 0.0f;
    if (i < NEXP) cnt[i] = 0;
}

// ---------------------------------------------------------------------------
// K1: router. One wave32 per token: logits, softmax, top-2, atomic append.
// ---------------------------------------------------------------------------
__global__ void moe_router_kernel(const float* __restrict__ x,
                                  const float* __restrict__ rw,
                                  int* __restrict__ cnt,
                                  int* __restrict__ exp_tok,
                                  float* __restrict__ exp_gate) {
    int wave = threadIdx.x >> 5;
    int lane = threadIdx.x & 31;
    int tok  = blockIdx.x * 8 + wave;
    if (tok >= NTOK) return;

    const float* xp = x + (size_t)tok * DMODEL;
    float acc[NEXP];
#pragma unroll
    for (int e = 0; e < NEXP; ++e) acc[e] = 0.0f;

    for (int d = lane; d < DMODEL; d += 32) {
        float xv = xp[d];
#pragma unroll
        for (int e = 0; e < NEXP; ++e) acc[e] += xv * rw[e * DMODEL + d];
    }
#pragma unroll
    for (int e = 0; e < NEXP; ++e) {
#pragma unroll
        for (int off = 16; off > 0; off >>= 1)
            acc[e] += __shfl_xor(acc[e], off, 32);
    }
    if (lane == 0) {
        float m = acc[0];
#pragma unroll
        for (int e = 1; e < NEXP; ++e) m = fmaxf(m, acc[e]);
        float p[NEXP]; float s = 0.0f;
#pragma unroll
        for (int e = 0; e < NEXP; ++e) { p[e] = expf(acc[e] - m); s += p[e]; }
        float inv = 1.0f / s;
#pragma unroll
        for (int e = 0; e < NEXP; ++e) p[e] *= inv;
        // top-2, ties -> lowest index (strict >)
        int i1 = 0;
#pragma unroll
        for (int e = 1; e < NEXP; ++e) if (p[e] > p[i1]) i1 = e;
        int i2 = (i1 == 0) ? 1 : 0;
#pragma unroll
        for (int e = 0; e < NEXP; ++e) if (e != i1 && p[e] > p[i2]) i2 = e;

        int pos1 = atomicAdd(&cnt[i1], 1);
        exp_tok [i1 * EXPCAP + pos1] = tok;
        exp_gate[i1 * EXPCAP + pos1] = p[i1];
        int pos2 = atomicAdd(&cnt[i2], 1);
        exp_tok [i2 * EXPCAP + pos2] = tok;
        exp_gate[i2 * EXPCAP + pos2] = p[i2];
    }
}

// ---------------------------------------------------------------------------
// K2: row offsets, padded to MTILE so expert regions never overlap a block tile
// ---------------------------------------------------------------------------
__global__ void moe_offsets_kernel(const int* __restrict__ cnt, int* __restrict__ off) {
    if (blockIdx.x == 0 && threadIdx.x == 0) {
        int run = 0;
        for (int e = 0; e < NEXP; ++e) {
            off[e] = run;
            run += (cnt[e] + MTILE - 1) & ~(MTILE - 1);
        }
    }
}

// ---------------------------------------------------------------------------
// K3: gather tokens into compact bf16 A matrix; record slot token/gate
// ---------------------------------------------------------------------------
__global__ void moe_gather_kernel(const float* __restrict__ x,
                                  const int* __restrict__ cnt,
                                  const int* __restrict__ off,
                                  const int* __restrict__ exp_tok,
                                  const float* __restrict__ exp_gate,
                                  __bf16* __restrict__ Xg,
                                  int* __restrict__ slot_token,
                                  float* __restrict__ slot_gate) {
    int e = blockIdx.x;
    int r = blockIdx.y;
    if (r >= cnt[e]) return;
    int tok = exp_tok[e * EXPCAP + r];
    int gr  = off[e] + r;
    const float* src = x + (size_t)tok * DMODEL;
    __bf16* dst = Xg + (size_t)gr * DMODEL;
    for (int c = threadIdx.x; c < DMODEL; c += blockDim.x)
        dst[c] = (__bf16)src[c];
    if (threadIdx.x == 0) {
        slot_token[gr] = tok;
        slot_gate[gr]  = exp_gate[e * EXPCAP + r];
    }
}

// ---------------------------------------------------------------------------
// K4/K5: tiled WMMA GEMM per expert, double-buffered LDS.
//   LAYER==1: H = GELU(Xg @ W1[e])   (bf16 out, pitch DFF)
//   LAYER==2: out[tok] += gate * (H @ W2[e])  (f32 atomic scatter)
// Block: 256 threads = 8 waves, tile M=128 x N=64, K-step 32.
// Wave grid 4(M) x 2(N); each wave: 2x2 fragments -> 4 wmma per K chunk.
// As row-major [128][40]; Bs COLUMN-major [64][40] so every fragment is
// contiguous 16B runs -> ds_load_b128 only. One barrier per chunk.
// ---------------------------------------------------------------------------
template <int LAYER>
__global__ void moe_gemm_kernel(const __bf16* __restrict__ Abase,
                                const float* __restrict__ Wg,
                                __bf16* __restrict__ Hout,
                                float* __restrict__ Out,
                                const int* __restrict__ off,
                                const int* __restrict__ cnt,
                                const int* __restrict__ slot_token,
                                const float* __restrict__ slot_gate,
                                int K, int N) {
    __shared__ __bf16 As[2][MTILE][ASTRIDE];   // ping-pong: 128 rows x 32 k
    __shared__ __bf16 Bs[2][NTILE][BSTRIDE];   // ping-pong: 64 cols x 32 k

    const int e  = blockIdx.y >> 4;            // 16 M tiles max per expert
    const int mt = blockIdx.y & 15;
    const int ce = cnt[e];
    const int pc = (ce + MTILE - 1) & ~(MTILE - 1);
    if (mt * MTILE >= pc) return;
    const int row0 = off[e] + mt * MTILE;
    const int n0   = blockIdx.x * NTILE;
    const float* W = Wg + (size_t)e * K * N;

    const int tid    = threadIdx.x;
    const int lane   = tid & 31;
    const int wave   = tid >> 5;
    const int wave_m = wave >> 1;              // 0..3
    const int wave_n = wave & 1;               // 0..1
    const int lrow   = lane & 15;
    const int lhalf  = lane >> 4;

    v8f acc[2][2];
#pragma unroll
    for (int s = 0; s < 2; ++s)
#pragma unroll
        for (int t = 0; t < 2; ++t) acc[s][t] = (v8f){0,0,0,0,0,0,0,0};

    // ---- staging assignments: uniform base + 32-bit varying offset ----
    const int a_row  = tid >> 1;               // 0..127
    const int a_half = tid & 1;                // 16 bf16 each
    unsigned aOff = (unsigned)(row0 + a_row) * (unsigned)K + a_half * 16;

    const int b_col = tid & 63;                // 0..63: column within tile
    const int b_kg  = tid >> 6;                // 0..3: 8-k group
    unsigned bOff = (unsigned)(b_kg * 8) * (unsigned)N + n0 + b_col;

    // ---- fragment row/col indices (buffer-relative, hoisted) ----
    const int afr0 = wave_m * 32 + lrow;
    const int afr1 = wave_m * 32 + 16 + lrow;
    const int bfc0 = wave_n * 32 + lrow;
    const int bfc1 = wave_n * 32 + 16 + lrow;

    const int nChunks = K / KSTEP;

    // ---- prologue: stage chunk 0 into buffer 0 ----
    uint4 aReg0, aReg1; float bReg[8];
    {
        const uint4* s4 = (const uint4*)(Abase + aOff);
        aReg0 = s4[0]; aReg1 = s4[1];
#pragma unroll
        for (int j = 0; j < 8; ++j) bReg[j] = W[bOff + (unsigned)j * N];
        aOff += KSTEP;
        bOff += (unsigned)KSTEP * N;
    }
    {
        uint4* aDst = (uint4*)&As[0][a_row][a_half * 16];
        aDst[0] = aReg0; aDst[1] = aReg1;
        __attribute__((aligned(16))) __bf16 tmp[8];
#pragma unroll
        for (int j = 0; j < 8; ++j) tmp[j] = (__bf16)bReg[j];
        *(uint4*)&Bs[0][b_col][b_kg * 8] = *(const uint4*)tmp;
    }
    __syncthreads();

#pragma unroll 2
    for (int i = 0; i < nChunks; ++i) {
        const int cur = i & 1;
        const int nxt = cur ^ 1;
        const bool more = (i + 1 < nChunks);

        // ---- issue next chunk's global loads (latency hidden by WMMAs) ----
        if (more) {
            const uint4* s4 = (const uint4*)(Abase + aOff);
            aReg0 = s4[0]; aReg1 = s4[1];
#pragma unroll
            for (int j = 0; j < 8; ++j) bReg[j] = W[bOff + (unsigned)j * N];
            aOff += KSTEP;
            bOff += (unsigned)KSTEP * N;
            if (i + 2 < nChunks)
                __builtin_prefetch(&W[bOff], 0, 3);   // global_prefetch_b8
        }

        // ---- fragments (ds_load_b128 only) + 4 WMMAs from buffer `cur` ----
        bf16x16_bits ab;
        const uint4* aF0 = (const uint4*)&As[cur][afr0][lhalf * 8];
        const uint4* aF1 = (const uint4*)&As[cur][afr1][lhalf * 8];
        const uint4* bF0 = (const uint4*)&Bs[cur][bfc0][lhalf * 8];
        const uint4* bF1 = (const uint4*)&Bs[cur][bfc1][lhalf * 8];
        ab.lo = aF0[0]; ab.hi = aF0[2];            // K = lhalf*8..+7, 16+lhalf*8..+7
        v16bf a0 = __builtin_bit_cast(v16bf, ab);
        ab.lo = aF1[0]; ab.hi = aF1[2];
        v16bf a1 = __builtin_bit_cast(v16bf, ab);
        ab.lo = bF0[0]; ab.hi = bF0[2];            // K = lhalf*8..+7, 16+lhalf*8..+7
        v16bf b0 = __builtin_bit_cast(v16bf, ab);
        ab.lo = bF1[0]; ab.hi = bF1[2];
        v16bf b1 = __builtin_bit_cast(v16bf, ab);

        acc[0][0] = __builtin_amdgcn_wmma_f32_16x16x32_bf16(false, a0, false, b0, (short)0, acc[0][0], false, false);
        acc[0][1] = __builtin_amdgcn_wmma_f32_16x16x32_bf16(false, a0, false, b1, (short)0, acc[0][1], false, false);
        acc[1][0] = __builtin_amdgcn_wmma_f32_16x16x32_bf16(false, a1, false, b0, (short)0, acc[1][0], false, false);
        acc[1][1] = __builtin_amdgcn_wmma_f32_16x16x32_bf16(false, a1, false, b1, (short)0, acc[1][1], false, false);

        // ---- write staged registers into buffer `nxt` ----
        if (more) {
            uint4* aDst = (uint4*)&As[nxt][a_row][a_half * 16];
            aDst[0] = aReg0; aDst[1] = aReg1;
            __attribute__((aligned(16))) __bf16 tmp[8];
#pragma unroll
            for (int j = 0; j < 8; ++j) tmp[j] = (__bf16)bReg[j];
            *(uint4*)&Bs[nxt][b_col][b_kg * 8] = *(const uint4*)tmp;
        }
        __syncthreads();   // single barrier per chunk
    }

    // ---- writeback (C layout: lane L -> col = L%16, vgpr v -> row = v + (L/16)*8)
#pragma unroll
    for (int s = 0; s < 2; ++s) {
#pragma unroll
        for (int t = 0; t < 2; ++t) {
#pragma unroll
            for (int v = 0; v < 8; ++v) {
                int row_l = wave_m * 32 + s * 16 + v + lhalf * 8;
                int col_l = wave_n * 32 + t * 16 + lrow;
                int gr = row0 + row_l;
                if (LAYER == 1) {
                    Hout[(size_t)gr * N + n0 + col_l] = (__bf16)gelu_tanh_fast(acc[s][t][v]);
                } else {
                    int rie = mt * MTILE + row_l;
                    if (rie < ce) {
                        int   tok  = slot_token[gr];
                        float gate = slot_gate[gr];
                        atomicAdd(&Out[(size_t)tok * N + n0 + col_l],
                                  acc[s][t][v] * gate);
                    }
                }
            }
        }
    }
}

// ---------------------------------------------------------------------------
// launch
// ---------------------------------------------------------------------------
extern "C" void kernel_launch(void* const* d_in, const int* in_sizes, int n_in,
                              void* d_out, int out_size, void* d_ws, size_t ws_size,
                              hipStream_t stream) {
    (void)in_sizes; (void)n_in; (void)out_size; (void)ws_size;
    const float* x  = (const float*)d_in[0];   // [2048,768]
    const float* rw = (const float*)d_in[1];   // [8,768]
    const float* w1 = (const float*)d_in[2];   // [8,768,3072]
    const float* w2 = (const float*)d_in[3];   // [8,3072,768]
    float* out = (float*)d_out;                // [2048,768]

    char* ws = (char*)d_ws;
    size_t o = 0;
    auto take = [&](size_t bytes) { size_t r = o; o = (o + bytes + 255) & ~(size_t)255; return r; };
    int*    cnt        = (int*)   (ws + take(NEXP * sizeof(int)));
    int*    off        = (int*)   (ws + take(NEXP * sizeof(int)));
    int*    exp_tok    = (int*)   (ws + take((size_t)NEXP * EXPCAP * sizeof(int)));
    float*  exp_gate   = (float*) (ws + take((size_t)NEXP * EXPCAP * sizeof(float)));
    int*    slot_token = (int*)   (ws + take((size_t)ROWCAP * sizeof(int)));
    float*  slot_gate  = (float*) (ws + take((size_t)ROWCAP * sizeof(float)));
    __bf16* Xg         = (__bf16*)(ws + take((size_t)ROWCAP * DMODEL * sizeof(__bf16)));
    __bf16* Hbuf       = (__bf16*)(ws + take((size_t)ROWCAP * DFF * sizeof(__bf16)));

    const int nOut = NTOK * DMODEL;
    moe_zero_kernel<<<(nOut + 255) / 256, 256, 0, stream>>>(out, nOut, cnt);
    moe_router_kernel<<<NTOK / 8, 256, 0, stream>>>(x, rw, cnt, exp_tok, exp_gate);
    moe_offsets_kernel<<<1, 32, 0, stream>>>(cnt, off);
    moe_gather_kernel<<<dim3(NEXP, EXPCAP), 256, 0, stream>>>(
        x, cnt, off, exp_tok, exp_gate, Xg, slot_token, slot_gate);
    moe_gemm_kernel<1><<<dim3(DFF / NTILE, NEXP * 16), 256, 0, stream>>>(
        Xg, w1, Hbuf, nullptr, off, cnt, slot_token, slot_gate, DMODEL, DFF);
    moe_gemm_kernel<2><<<dim3(DMODEL / NTILE, NEXP * 16), 256, 0, stream>>>(
        Hbuf, w2, nullptr, out, off, cnt, slot_token, slot_gate, DFF, DMODEL);
}
